// SparseLossFeature_481036337369
// MI455X (gfx1250) — matrix operand order repaired and checked by reading, not compiled, not run on hardware
//
#include <hip/hip_runtime.h>
#include <hip/hip_bf16.h>
#include <math.h>

typedef __attribute__((ext_vector_type(16))) _Float16 v16h;
typedef __attribute__((ext_vector_type(8)))  float    v8f;
typedef __attribute__((ext_vector_type(4)))  unsigned v4u;
typedef __attribute__((ext_vector_type(8)))  int      v8i;
typedef __attribute__((ext_vector_type(4)))  int      v4i;

#define N_SAMPLES 32768
#define N_FILTERS 64
#define FILTER_SIZE 64
#define N_BANDS 7
#define N_KEEP 128
#define NSIG 16          // 8 target + 8 recon

#if defined(__has_builtin)
#if __has_builtin(__builtin_amdgcn_tensor_load_to_lds)
#define HAVE_TDM 1
#endif
#endif

// ---------- order-preserving float <-> uint encoding (for atomic max-pool) ----
__device__ __forceinline__ unsigned encodeOrd(float v){
  unsigned b = __float_as_uint(v);
  return (b & 0x80000000u) ? ~b : (b | 0x80000000u);
}
__device__ __forceinline__ float decodeOrd(unsigned u){
  unsigned b = (u & 0x80000000u) ? (u & 0x7fffffffu) : ~u;
  return __uint_as_float(b);
}

// ---------- init: pooled accumulators to 0 (== below any real value), out=0 ---
__global__ void init_kernel(unsigned* __restrict__ pooled, float* __restrict__ out, int n){
  int i = blockIdx.x * blockDim.x + threadIdx.x;
  if (i < n) pooled[i] = 0u;
  if (i == 0) out[0] = 0.f;
}

// ---------- filters -> f16 -------------------------------------------------
__global__ void filt_convert_kernel(const float* __restrict__ filt, _Float16* __restrict__ f16){
  int i = blockIdx.x * blockDim.x + threadIdx.x;
  if (i < N_FILTERS * FILTER_SIZE) f16[i] = (_Float16)filt[i];
}

// ---------- forward 32768-pt FFT, whole transform resident in 256KB LDS -----
// One workgroup per signal; writes ortho-normalized rfft bins [0..16384].
__global__ void fft_fwd_kernel(const float* __restrict__ tgt,
                               const float* __restrict__ rec,
                               float2* __restrict__ spec){
  extern __shared__ float2 sm[];
  const int s   = blockIdx.x;
  const int tid = threadIdx.x;
  const float* x = (s < 8) ? (tgt + (size_t)s * N_SAMPLES)
                           : (rec + (size_t)(s - 8) * N_SAMPLES);
  for (int i = tid; i < N_SAMPLES; i += blockDim.x){
    unsigned r = __brev((unsigned)i) >> 17;           // 15-bit bit-reverse
    sm[r] = make_float2(x[i], 0.f);
  }
  __syncthreads();
  for (int len = 2; len <= N_SAMPLES; len <<= 1){
    int half = len >> 1;
    for (int j = tid; j < N_SAMPLES / 2; j += blockDim.x){
      int blk = j / half, k = j - blk * half;
      int i0 = blk * len + k, i1 = i0 + half;
      float ang = -6.283185307179586f * (float)k / (float)len;
      float sn, cs; __sincosf(ang, &sn, &cs);
      float2 a = sm[i0], b = sm[i1];
      float2 t = make_float2(b.x * cs - b.y * sn, b.x * sn + b.y * cs);
      sm[i0] = make_float2(a.x + t.x, a.y + t.y);
      sm[i1] = make_float2(a.x - t.x, a.y - t.y);
    }
    __syncthreads();
  }
  const float sc = 1.0f / sqrtf((float)N_SAMPLES);    // norm='ortho'
  for (int q = tid; q <= N_SAMPLES / 2; q += blockDim.x)
    spec[(size_t)s * 16400 + q] = make_float2(sm[q].x * sc, sm[q].y * sc);
}

// ---------- per-band Hermitian inverse FFT -> padded f16 band signal --------
__global__ void ifft_band_kernel(const float2* __restrict__ spec,
                                 _Float16* __restrict__ bands16){
  extern __shared__ float2 sm[];
  const int s = blockIdx.x, d = blockIdx.y, tid = threadIdx.x;
  const int M     = 512 << d;
  const int half  = M >> 1;
  const int start = (d == 0) ? 0 : (128 << d);
  const int end   = 256 << d;
  const int lg    = 9 + d;
  const float2* sp = spec + (size_t)s * 16400;
  for (int q = tid; q <= half; q += blockDim.x){
    float2 v = make_float2(0.f, 0.f);
    if (q >= start && q < end) v = sp[q];
    sm[__brev((unsigned)q) >> (32 - lg)] = v;
    if (q >= 1 && q < half)
      sm[__brev((unsigned)(M - q)) >> (32 - lg)] = make_float2(v.x, -v.y);
  }
  __syncthreads();
  for (int len = 2; len <= M; len <<= 1){
    int h = len >> 1;
    for (int j = tid; j < (M >> 1); j += blockDim.x){
      int blk = j / h, k = j - blk * h;
      int i0 = blk * len + k, i1 = i0 + h;
      float ang = 6.283185307179586f * (float)k / (float)len;  // inverse
      float sn, cs; __sincosf(ang, &sn, &cs);
      float2 a = sm[i0], b = sm[i1];
      float2 t = make_float2(b.x * cs - b.y * sn, b.x * sn + b.y * cs);
      sm[i0] = make_float2(a.x + t.x, a.y + t.y);
      sm[i1] = make_float2(a.x - t.x, a.y - t.y);
    }
    __syncthreads();
  }
  const float sc = 1.0f / sqrtf((float)M);            // ortho inverse scale
  const int pd = 64 * d + 512 * ((1 << d) - 1);       // padded band offset
  _Float16* dst = bands16 + (size_t)s * 65536 + pd;
  for (int i = tid; i < 64; i += blockDim.x) dst[i] = (_Float16)0.f;
  for (int t = tid; t < M; t += blockDim.x) dst[64 + t] = (_Float16)(sm[t].x * sc);
}

// ---------- 64-tap FIR as WMMA GEMM (M=64 filters, K=64 taps) + fused pool --
// Filters staged into LDS by the Tensor Data Mover; band window staged in LDS.
__global__ void conv_pool_wmma_kernel(const _Float16* __restrict__ bands16,
                                      const _Float16* __restrict__ filt16,
                                      unsigned* __restrict__ pooled,
                                      int d){
  __shared__ _Float16 lds_filt[N_FILTERS * FILTER_SIZE];  // 8KB, reused by all tiles
  __shared__ _Float16 lds_band[128];                      // chunk window [t0c-64, t0c+64)
  const int s     = blockIdx.y;
  const int chunk = blockIdx.x;        // 64 output samples per workgroup
  const int lane  = threadIdx.x & 31;
  const int wave  = threadIdx.x >> 5;  // 8 waves
  const int stride = 8 << d;
  const int pad    = stride >> 1;
  const int pd = 64 * d + 512 * ((1 << d) - 1);
  const _Float16* band = bands16 + (size_t)s * 65536 + pd + 64;  // band[t], t>=-64 ok
  const int t0c = chunk * 64;
  const int hl = lane >> 4;            // lane half (0/1)
  const int l  = lane & 15;
  unsigned* pool = pooled + (size_t)(s * N_BANDS + d) * 4096;

  // --- stage filter matrix via TDM (async tensor DMA, TENSORcnt) ------------
#ifdef HAVE_TDM
  if (threadIdx.x < 32){
    unsigned lds_off = (unsigned)(size_t)(void*)lds_filt;         // LDS byte addr
    unsigned long long ga = (unsigned long long)(size_t)filt16;   // 57-bit global
    v4u g0; // D# group 0: count=1 | lds_addr | global_addr | type=2
    g0[0] = 1u;
    g0[1] = lds_off;
    g0[2] = (unsigned)(ga & 0xffffffffu);
    g0[3] = (unsigned)((ga >> 32) & 0x01ffffffu) | (2u << 30);
    v8i g1; // D# group 1: data_size=2B, 64x64 tensor, 64x64 tile, stride 64
    g1[0] = (int)(1u << 16);    // data_size=1 (2 bytes); no multicast/pad/iter
    g1[1] = (int)(64u << 16);   // tensor_dim0 = 64  (bits 79:48)
    g1[2] = (int)(64u << 16);   // tensor_dim1 = 64  (bits 111:80)
    g1[3] = (int)(64u << 16);   // tile_dim0   = 64  (bits 127:112)
    g1[4] = 64;                 // tile_dim1   = 64, tile_dim2 = 0
    g1[5] = 64;                 // tensor_dim0_stride = 64
    g1[6] = 0;                  // stride hi | tensor_dim1_stride lo
    g1[7] = 0;
    v4i g2 = {0, 0, 0, 0};
    v4i g3 = {0, 0, 0, 0};
#if __clang_major__ >= 23
    v8i gz = {0, 0, 0, 0, 0, 0, 0, 0};
    __builtin_amdgcn_tensor_load_to_lds(g0, g1, g2, g3, gz, 0);
#else
    __builtin_amdgcn_tensor_load_to_lds(g0, g1, g2, g3, 0);
#endif
    __builtin_amdgcn_s_wait_tensorcnt(0);
  }
#else
  for (int i = threadIdx.x; i < N_FILTERS * FILTER_SIZE; i += 256)
    lds_filt[i] = filt16[i];
#endif
  // --- stage band window: 128 halves covering every t-k this chunk needs ----
  if (threadIdx.x < 128)
    lds_band[threadIdx.x] = band[t0c - 64 + (int)threadIdx.x];
  // prefetch next chunk's window into cache (global_prefetch_b8)
  __builtin_prefetch((const void*)(band + t0c + 64), 0, 1);
  __syncthreads();

  #pragma unroll
  for (int rep = 0; rep < 2; ++rep){
    int tile = wave + rep * 8;         // 16 tiles: 4 M-tiles x 4 N-tiles
    int m0 = (tile & 3) * 16;
    int nt = (tile >> 2) * 16;         // local time base within chunk
    v8f c = {};
    #pragma unroll
    for (int ks = 0; ks < 64; ks += 32){
      v16h a, b;
      #pragma unroll
      for (int e = 0; e < 16; ++e){
        // A (16x32, ISA 7.12.2): lane<16 rows hold K {0..7,16..23}, lane>=16 {8..15,24..31}
        int ka = hl * 8 + e + ((e >= 8) ? 8 : 0);
        a[e] = lds_filt[(m0 + l) * FILTER_SIZE + ks + ka];
        // B (32x16): lanes 0-15 hold K {0..7,16..23}, lanes 16-31 {8..15,24..31}; column = lane&15
        int kb = hl * 8 + e + ((e >= 8) ? 8 : 0);
        b[e] = lds_band[64 + nt + l - (ks + kb)];
      }
      c = __builtin_amdgcn_wmma_f32_16x16x32_f16(false, a, false, b,
                                                 (short)0, c, false, false);
    }
    // C/D layout: VGPR r, lane<16 -> row m0+r, lane>=16 -> row m0+r+8; col = lane&15
    #pragma unroll
    for (int r = 0; r < 8; ++r){
      int f = m0 + r + hl * 8;
      int t = t0c + nt + l;
      unsigned u = encodeOrd(c[r]);
      int j1 = (t + pad) / stride;     // windows j covering t: {j1-1, j1} clipped
      #pragma unroll
      for (int dj = -1; dj <= 0; ++dj){
        int j = j1 + dj;
        if (j >= 0 && j < 64){
          int lo = j * stride - pad;
          if (t >= lo && t < lo + 2 * stride)
            atomicMax(&pool[f * 64 + j], u);
        }
      }
    }
  }
}

// ---------- top-128 per (signal, band): iterative argmax (matches top_k order)
__global__ void topk_kernel(const unsigned* __restrict__ pooled,
                            float* __restrict__ tkval,
                            unsigned* __restrict__ tkch,
                            unsigned* __restrict__ tktm){
  __shared__ float vals[4096];
  __shared__ float rv[256];
  __shared__ int   ri[256];
  const int s = blockIdx.x, d = blockIdx.y, tid = threadIdx.x;
  const unsigned* pool = pooled + (size_t)(s * N_BANDS + d) * 4096;
  for (int i = tid; i < 4096; i += 256) vals[i] = decodeOrd(pool[i]);
  __syncthreads();
  const size_t obase = (size_t)(s * N_BANDS + d) * N_KEEP;
  for (int k = 0; k < N_KEEP; ++k){
    float best = -INFINITY; int bi = 0x7fffffff;
    for (int i = tid * 16; i < tid * 16 + 16; ++i){
      float v = vals[i];
      if (v > best){ best = v; bi = i; }     // strict > keeps lowest index on ties
    }
    rv[tid] = best; ri[tid] = bi;
    __syncthreads();
    for (int off = 128; off > 0; off >>= 1){
      if (tid < off){
        float ov = rv[tid + off]; int oi = ri[tid + off];
        if (ov > rv[tid] || (ov == rv[tid] && oi < ri[tid])){ rv[tid] = ov; ri[tid] = oi; }
      }
      __syncthreads();
    }
    if (tid == 0){
      int idx = ri[0];
      tkval[obase + k] = rv[0];
      tkch [obase + k] = (unsigned)(idx >> 6);   // flat = ch*64 + tm
      tktm [obase + k] = (unsigned)(idx & 63);
      vals[idx] = -INFINITY;
    }
    __syncthreads();
  }
}

// ---------- feature gather + |t - r| mean reduction -------------------------
__global__ void loss_kernel(const float* __restrict__ tkval,
                            const unsigned* __restrict__ tkch,
                            const unsigned* __restrict__ tktm,
                            const float* __restrict__ proj_time,
                            const float* __restrict__ proj_freq,
                            float* __restrict__ out){
  __shared__ float red[256];
  const int b = blockIdx.x, d = blockIdx.y, k = blockIdx.z, tid = threadIdx.x;
  size_t ti = (size_t)(b * N_BANDS + d) * N_KEEP + k;
  size_t rj = (size_t)((b + 8) * N_BANDS + d) * N_KEEP + k;
  float    tval = tkval[ti], rval = tkval[rj];
  unsigned ttm  = tktm[ti],  rtm  = tktm[rj];
  unsigned tch  = tkch[ti],  rch  = tkch[rj];
  float v;
  if (tid < 128){
    int j = tid;
    v = fabsf(tval * proj_time[ttm * 128 + j] - rval * proj_time[rtm * 128 + j]);
  } else {
    int j = tid - 128;
    v = fabsf(proj_freq[tch * 128 + j] - proj_freq[rch * 128 + j]);
  }
  red[tid] = v;
  __syncthreads();
  for (int off = 128; off > 0; off >>= 1){
    if (tid < off) red[tid] += red[tid + off];
    __syncthreads();
  }
  if (tid == 0) atomicAdd(out, red[0] * (1.0f / 1835008.0f));  // mean over 8*128*1792
}

extern "C" void kernel_launch(void* const* d_in, const int* in_sizes, int n_in,
                              void* d_out, int out_size, void* d_ws, size_t ws_size,
                              hipStream_t stream){
  (void)in_sizes; (void)n_in; (void)out_size; (void)ws_size;
  const float* tgt       = (const float*)d_in[0];
  const float* rec       = (const float*)d_in[1];
  const float* filt      = (const float*)d_in[2];
  const float* proj_time = (const float*)d_in[3];
  const float* proj_freq = (const float*)d_in[4];
  char* ws = (char*)d_ws;
  float2*   spec    = (float2*)  (ws);             // 16*16400 float2 = 2,099,200 B
  _Float16* bands16 = (_Float16*)(ws + 2099200);   // 16*65536 halves = 2,097,152 B
  _Float16* filt16  = (_Float16*)(ws + 4196352);   // 4096 halves     =     8,192 B
  unsigned* pooled  = (unsigned*)(ws + 4204544);   // 16*7*4096 u32   = 1,835,008 B
  float*    tkval   = (float*)   (ws + 6039552);   // 16*7*128 f32
  unsigned* tkch    = (unsigned*)(ws + 6096896);
  unsigned* tktm    = (unsigned*)(ws + 6154240);
  float* out = (float*)d_out;

  const int npool = NSIG * N_BANDS * 4096;
  init_kernel<<<(npool + 255) / 256, 256, 0, stream>>>(pooled, out, npool);
  filt_convert_kernel<<<(N_FILTERS * FILTER_SIZE + 255) / 256, 256, 0, stream>>>(filt, filt16);
  // 256 KB dynamic LDS: whole 32k-pt FFT resident per WGP (CDNA5: 320KB LDS/WGP)
  fft_fwd_kernel<<<NSIG, 256, N_SAMPLES * sizeof(float2), stream>>>(tgt, rec, spec);
  ifft_band_kernel<<<dim3(NSIG, N_BANDS), 256, N_SAMPLES * sizeof(float2), stream>>>(spec, bands16);
  for (int d = 0; d < N_BANDS; ++d){
    dim3 grid(8 << d, NSIG);
    conv_pool_wmma_kernel<<<grid, 256, 0, stream>>>(bands16, filt16, pooled, d);
  }
  topk_kernel<<<dim3(NSIG, N_BANDS), 256, 0, stream>>>(pooled, tkval, tkch, tktm);
  loss_kernel<<<dim3(8, N_BANDS, N_KEEP), 256, 0, stream>>>(tkval, tkch, tktm,
                                                            proj_time, proj_freq, out);
}
